// PrunableGRU_23648089931978
// MI455X (gfx1250) — compile-verified, matrix operand order checked by reading, not run on hardware
//
#include <hip/hip_runtime.h>
#include <hip/hip_bf16.h>
#include <stdint.h>

// Problem sizes (match reference)
#define S_LEN   1024
#define IN_DIM  2048
#define H_DIM   2048
#define H3      (3 * H_DIM)
#define NB_REC  64            // persistent blocks for the recurrent scan

typedef __attribute__((ext_vector_type(16))) __bf16 v16bf;
typedef __attribute__((ext_vector_type(8)))  float  v8f;

// ---------- small helpers ----------------------------------------------------

__device__ __forceinline__ unsigned short f2bf_rne(float f) {
  unsigned u = __float_as_uint(f);
  unsigned r = (u + 0x7fffu + ((u >> 16) & 1u)) >> 16;
  return (unsigned short)r;
}
__device__ __forceinline__ float bflo(unsigned u) { return __uint_as_float(u << 16); }
__device__ __forceinline__ float bfhi(unsigned u) { return __uint_as_float(u & 0xffff0000u); }

__global__ void cvt_f32_bf16(const float* __restrict__ in, unsigned short* __restrict__ out, int n) {
  int i = blockIdx.x * blockDim.x + threadIdx.x;
  if (i < n) out[i] = f2bf_rne(in[i]);
}

// Tiled transpose + convert: out[c*R + r] = bf16(in[r*C + c])
__global__ void __launch_bounds__(256)
cvtT_f32_bf16(const float* __restrict__ in, unsigned short* __restrict__ out, int R, int C) {
  __shared__ float t[32][33];
  int bx = blockIdx.x * 32;   // col base
  int by = blockIdx.y * 32;   // row base
  int tx = threadIdx.x & 31;
  int ty = threadIdx.x >> 5;  // 0..7
#pragma unroll
  for (int j = 0; j < 4; j++)
    t[ty + 8 * j][tx] = in[(size_t)(by + ty + 8 * j) * C + bx + tx];
  __syncthreads();
#pragma unroll
  for (int j = 0; j < 4; j++)
    out[(size_t)(bx + ty + 8 * j) * R + by + tx] = f2bf_rne(t[tx][ty + 8 * j]);
}

// ---------- CDNA5 async global->LDS staging ---------------------------------
// GLOBAL_LOAD_ASYNC_TO_LDS_B128: per-lane 16B  LDS[vdst] = MEM[vaddr];
// tracked by ASYNCcnt (loads complete in issue order -> partial waits work).

__device__ __forceinline__ void async_g2l_b128(unsigned lds_off, const void* gaddr) {
  asm volatile("global_load_async_to_lds_b128 %0, %1, off"
               :: "v"(lds_off), "v"(gaddr) : "memory");
}
__device__ __forceinline__ void wait_async_le4() {
  asm volatile("s_wait_asynccnt 0x4" ::: "memory");
}
__device__ __forceinline__ void wait_async_0() {
  asm volatile("s_wait_asynccnt 0x0" ::: "memory");
}

// ---------- WMMA fragment loader (works for LDS or global bf16) --------------
// 16-bit A (16x32) lane layout (ISA 7.12.2): lane<16 holds row m=lane with
// K {0..7, 16..23}; lane>=16 holds row m=lane-16 with K {8..15, 24..31}.
// B (32x16) symmetric with lanes holding columns (N x K storage).

__device__ __forceinline__ v16bf frag_rowK(const unsigned short* __restrict__ base,
                                           int ld, int row0, int lane) {
  const unsigned short* p = base + (size_t)(row0 + (lane & 15)) * ld + ((lane >> 4) << 3);
  union { v16bf v; uint4 q[2]; } u;
  u.q[0] = *(const uint4*)(p);
  u.q[1] = *(const uint4*)(p + 16);
  return u.v;
}

__device__ __forceinline__ v8f wmma_bf16(v16bf a, v16bf b, v8f c) {
  return __builtin_amdgcn_wmma_f32_16x16x32_bf16(false, a, false, b, (short)0, c, false, false);
}

// ---------- staged WMMA GEMM -------------------------------------------------
// C(MxN) = A(bf16, MxK row-major) @ B(bf16, NxK row-major)^T  [+ bias]
// Block tile 128x128, 8 waves; wave w owns rows [w*16, w*16+16) x all 128 cols
// (8 accumulators -> 8 wmma per A fragment). A/B 128x32 tiles double-buffered
// in LDS, filled with async b128 copies (4 per thread per stage). All 8 B
// fragments are preloaded so the 16 ds_loads clause together and the 8 WMMAs
// issue back-to-back.

template <int BF16_OUT>
__global__ void __launch_bounds__(256)
gemm_staged(const unsigned short* __restrict__ A,
            const unsigned short* __restrict__ B,
            const float* __restrict__ bias,
            void* __restrict__ Cout, int M, int N, int K) {
  __shared__ __align__(16) unsigned short As[2][128 * 32];
  __shared__ __align__(16) unsigned short Bs[2][128 * 32];

  const int tid  = threadIdx.x;
  const int lane = tid & 31;
  const int w    = tid >> 5;                 // 0..7
  const int m0   = blockIdx.y * 128;
  const int n0   = blockIdx.x * 128;

  const unsigned short* Ag = A + (size_t)m0 * K;   // advance by k below
  const unsigned short* Bg = B + (size_t)n0 * K;

  // stage tile (k0) into buffer bb: 128x32 of A and B, 512 b128 xfers each
  auto stage = [&](int bb, int k0) {
#pragma unroll
    for (int j = 0; j < 2; j++) {
      int id  = tid + j * 256;
      int row = id >> 2;
      int ch  = (id & 3) * 8;                // 8 ushorts = 16B chunk
      async_g2l_b128((unsigned)(uintptr_t)&As[bb][row * 32 + ch],
                     Ag + (size_t)row * K + k0 + ch);
      async_g2l_b128((unsigned)(uintptr_t)&Bs[bb][row * 32 + ch],
                     Bg + (size_t)row * K + k0 + ch);
    }
  };

  v8f acc[8];
#pragma unroll
  for (int nt = 0; nt < 8; nt++) acc[nt] = (v8f){};

  const int KT = K / 32;
  stage(0, 0);
  for (int kt = 0; kt < KT; kt++) {
    int bb = kt & 1;
    if (kt + 1 < KT) {          // prefetch next tile, then wait only the old one
      stage(bb ^ 1, (kt + 1) * 32);
      wait_async_le4();
    } else {
      wait_async_0();
    }
    __syncthreads();            // stage(kt) visible to all waves

    // Preload A + all 8 B fragments (independent ds_loads -> one clause),
    // then run the 8 WMMAs back-to-back.
    v16bf a = frag_rowK(&As[bb][0], 32, w * 16, lane);
    v16bf bf[8];
#pragma unroll
    for (int nt = 0; nt < 8; nt++)
      bf[nt] = frag_rowK(&Bs[bb][0], 32, nt * 16, lane);
#pragma unroll
    for (int nt = 0; nt < 8; nt++)
      acc[nt] = wmma_bf16(a, bf[nt], acc[nt]);

    __syncthreads();            // everyone done reading bb before it refills
  }

  // C/D layout: VGPR r -> M = 16*w + r + 8*(lane>=16); N = nt*16 + (lane&15)
  const int half = lane >> 4;
#pragma unroll
  for (int nt = 0; nt < 8; nt++) {
    int col = n0 + nt * 16 + (lane & 15);
    float bb = BF16_OUT ? 0.f : bias[col];
#pragma unroll
    for (int r = 0; r < 8; r++) {
      int row = m0 + w * 16 + half * 8 + r;
      if (BF16_OUT)
        ((unsigned short*)Cout)[(size_t)row * N + col] = f2bf_rne(acc[nt][r]);
      else
        ((float*)Cout)[(size_t)row * N + col] = acc[nt][r] + bb;
    }
  }
}

// ---------- b_comb[j] = b_hh[j] + sum_k W_hh[j,k] * b_hm[k] -----------------

__global__ void __launch_bounds__(256)
bcomb_kernel(const float* __restrict__ Whh, const float* __restrict__ bhh,
             const float* __restrict__ bhm, float* __restrict__ bcomb) {
  int lane = threadIdx.x & 31;
  int w    = threadIdx.x >> 5;
  int j    = blockIdx.x * 8 + w;
  float acc = 0.f;
  for (int k = lane; k < H_DIM; k += 32)
    acc += Whh[(size_t)j * H_DIM + k] * bhm[k];
#pragma unroll
  for (int off = 16; off; off >>= 1) acc += __shfl_xor(acc, off, 32);
  if (lane == 0) bcomb[j] = acc + bhh[j];
}

// ---------- grid-wide barrier (persistent scan kernel) ----------------------

struct GBar { unsigned cnt; unsigned gen; };

__global__ void bar_init(GBar* b) { b->cnt = 0u; b->gen = 0u; }

__device__ __forceinline__ void grid_sync(GBar* b, unsigned nb) {
  __syncthreads();
  if (threadIdx.x == 0) {
    __threadfence();
    unsigned g   = __hip_atomic_load(&b->gen, __ATOMIC_RELAXED, __HIP_MEMORY_SCOPE_AGENT);
    unsigned old = atomicAdd(&b->cnt, 1u);
    if (old == nb - 1u) {
      b->cnt = 0u;
      __threadfence();
      atomicAdd(&b->gen, 1u);
    } else {
      while (__hip_atomic_load(&b->gen, __ATOMIC_RELAXED, __HIP_MEMORY_SCOPE_AGENT) == g)
        __builtin_amdgcn_s_sleep(2);
    }
  }
  __syncthreads();
}

// ---------- persistent recurrent scan ---------------------------------------
// gh = h_prev @ (W_hh W_hm)^T + b_comb, so all per-step quantities depend only
// on h_prev -> one grid sync per timestep. Weights stream from L2 (both
// layers' recurrent weights fit in 192MB L2 in bf16).

__global__ void __launch_bounds__(256)
gru_scan(const float* __restrict__ gi,            // S x 3H, f32
         const unsigned short* __restrict__ Wc,   // 3H x H, bf16
         const unsigned short* __restrict__ Whm,  // H x H, bf16
         const float* __restrict__ bcomb,         // 3H
         const float* __restrict__ bhm,           // H
         const float* __restrict__ h0,            // H
         float* __restrict__ y,                   // S x H out sequence
         float* __restrict__ hlast,               // H
         float* hbuf,                             // 2 x H double buffer
         GBar* bar) {
  __shared__ float hsh[H_DIM];
  const int tid  = threadIdx.x;
  const int lane = tid & 31;
  const int gw   = blockIdx.x * 8 + (tid >> 5);   // 0..511

  if (blockIdx.x == 0)
    for (int i = tid; i < H_DIM; i += 256) hbuf[i] = h0[i];
  grid_sync(bar, NB_REC);

  for (int t = 0; t < S_LEN; t++) {
    const float* hprev = hbuf + (t & 1) * H_DIM;
    for (int i = tid; i < H_DIM; i += 256) hsh[i] = hprev[i];
    __syncthreads();
    float* hnext = hbuf + ((t + 1) & 1) * H_DIM;
    const float* git = gi + (size_t)t * H3;
    if (t + 1 < S_LEN) __builtin_prefetch(git + H3 + tid * 8, 0, 0);

    for (int q = 0; q < 4; q++) {
      const int i = gw * 4 + q;                       // 0..2047
      const unsigned short* rr = Wc  + (size_t)i * H_DIM;
      const unsigned short* rz = Wc  + (size_t)(H_DIM + i) * H_DIM;
      const unsigned short* rn = Wc  + (size_t)(2 * H_DIM + i) * H_DIM;
      const unsigned short* rm = Whm + (size_t)i * H_DIM;
      float s0 = 0.f, s1 = 0.f, s2 = 0.f, s3 = 0.f;
#pragma unroll 4
      for (int j = 0; j < H_DIM / 128; j++) {         // 16 iters, 4 elems/lane
        int k = (j * 32 + lane) * 4;
        float4 hv = *(const float4*)&hsh[k];
        uint2 w0 = *(const uint2*)(rr + k);
        uint2 w1 = *(const uint2*)(rz + k);
        uint2 w2 = *(const uint2*)(rn + k);
        uint2 w3 = *(const uint2*)(rm + k);
        s0 += hv.x * bflo(w0.x) + hv.y * bfhi(w0.x) + hv.z * bflo(w0.y) + hv.w * bfhi(w0.y);
        s1 += hv.x * bflo(w1.x) + hv.y * bfhi(w1.x) + hv.z * bflo(w1.y) + hv.w * bfhi(w1.y);
        s2 += hv.x * bflo(w2.x) + hv.y * bfhi(w2.x) + hv.z * bflo(w2.y) + hv.w * bfhi(w2.y);
        s3 += hv.x * bflo(w3.x) + hv.y * bfhi(w3.x) + hv.z * bflo(w3.y) + hv.w * bfhi(w3.y);
      }
#pragma unroll
      for (int off = 16; off; off >>= 1) {
        s0 += __shfl_xor(s0, off, 32);
        s1 += __shfl_xor(s1, off, 32);
        s2 += __shfl_xor(s2, off, 32);
        s3 += __shfl_xor(s3, off, 32);
      }
      if (lane == 0) {
        float r  = 1.f / (1.f + __expf(-(s0 + bcomb[i] + git[i])));
        float z  = 1.f / (1.f + __expf(-(s1 + bcomb[H_DIM + i] + git[H_DIM + i])));
        float hm = s3 + bhm[i];
        float n  = tanhf(git[2 * H_DIM + i] + r * (s2 + bcomb[2 * H_DIM + i]));
        float hn = (1.f - z) * n + z * hm;
        hnext[i] = hn;
        y[(size_t)t * H_DIM + i] = hn;
      }
    }
    grid_sync(bar, NB_REC);
  }

  if (blockIdx.x == 0)
    for (int i = tid; i < H_DIM; i += 256) hlast[i] = hbuf[(S_LEN & 1) * H_DIM + i];
}

// ---------- host orchestration ----------------------------------------------

extern "C" void kernel_launch(void* const* d_in, const int* in_sizes, int n_in,
                              void* d_out, int out_size, void* d_ws, size_t ws_size,
                              hipStream_t stream) {
  (void)in_sizes; (void)n_in; (void)out_size; (void)ws_size;

  const float* x    = (const float*)d_in[0];
  const float* hx   = (const float*)d_in[1];
  const float* Wih0 = (const float*)d_in[2];
  const float* bih0 = (const float*)d_in[3];
  const float* Whh0 = (const float*)d_in[4];
  const float* bhh0 = (const float*)d_in[5];
  const float* Whm0 = (const float*)d_in[6];
  const float* bhm0 = (const float*)d_in[7];
  const float* Wih1 = (const float*)d_in[8];
  const float* bih1 = (const float*)d_in[9];
  const float* Whh1 = (const float*)d_in[10];
  const float* bhh1 = (const float*)d_in[11];
  const float* Whm1 = (const float*)d_in[12];
  const float* bhm1 = (const float*)d_in[13];
  float* out = (float*)d_out;   // [S*H] out seq  |  [L*H] h_n

  size_t off = 0;
  auto alloc = [&](size_t bytes) -> void* {
    void* r = (char*)d_ws + off;
    off = (off + bytes + 255) & ~(size_t)255;
    return r;
  };
  unsigned short* xbf   = (unsigned short*)alloc((size_t)S_LEN * IN_DIM * 2);
  unsigned short* wih0b = (unsigned short*)alloc((size_t)H3 * IN_DIM * 2);
  unsigned short* wih1b = (unsigned short*)alloc((size_t)H3 * H_DIM * 2);
  unsigned short* whh0b = (unsigned short*)alloc((size_t)H3 * H_DIM * 2);
  unsigned short* whh1b = (unsigned short*)alloc((size_t)H3 * H_DIM * 2);
  unsigned short* whm0b = (unsigned short*)alloc((size_t)H_DIM * H_DIM * 2);
  unsigned short* whm1b = (unsigned short*)alloc((size_t)H_DIM * H_DIM * 2);
  unsigned short* whm0T = (unsigned short*)alloc((size_t)H_DIM * H_DIM * 2);
  unsigned short* whm1T = (unsigned short*)alloc((size_t)H_DIM * H_DIM * 2);
  unsigned short* wc0   = (unsigned short*)alloc((size_t)H3 * H_DIM * 2);
  unsigned short* wc1   = (unsigned short*)alloc((size_t)H3 * H_DIM * 2);
  float* bc0  = (float*)alloc((size_t)H3 * 4);
  float* bc1  = (float*)alloc((size_t)H3 * 4);
  float* gi   = (float*)alloc((size_t)S_LEN * H3 * 4);
  float* y0   = (float*)alloc((size_t)S_LEN * H_DIM * 4);
  unsigned short* y0b = (unsigned short*)alloc((size_t)S_LEN * H_DIM * 2);
  float* hbuf = (float*)alloc((size_t)2 * H_DIM * 4);
  GBar* bar   = (GBar*)alloc(256);

  auto cvt = [&](const float* src, unsigned short* dst, int n) {
    cvt_f32_bf16<<<(n + 255) / 256, 256, 0, stream>>>(src, dst, n);
  };

  // bf16 (and transposed-bf16) copies of GEMM operands
  cvt(x,    xbf,   S_LEN * IN_DIM);
  cvt(Wih0, wih0b, H3 * IN_DIM);
  cvt(Wih1, wih1b, H3 * H_DIM);
  cvt(Whh0, whh0b, H3 * H_DIM);
  cvt(Whh1, whh1b, H3 * H_DIM);
  cvt(Whm0, whm0b, H_DIM * H_DIM);
  cvt(Whm1, whm1b, H_DIM * H_DIM);
  dim3 gT(H_DIM / 32, H_DIM / 32);
  cvtT_f32_bf16<<<gT, 256, 0, stream>>>(Whm0, whm0T, H_DIM, H_DIM);
  cvtT_f32_bf16<<<gT, 256, 0, stream>>>(Whm1, whm1T, H_DIM, H_DIM);

  // W_comb = W_hh @ W_hm  (B operand = W_hm^T in NxK form); b_comb
  dim3 gcomb(H_DIM / 128, H3 / 128);
  gemm_staged<1><<<gcomb, 256, 0, stream>>>(whh0b, whm0T, nullptr, wc0, H3, H_DIM, H_DIM);
  gemm_staged<1><<<gcomb, 256, 0, stream>>>(whh1b, whm1T, nullptr, wc1, H3, H_DIM, H_DIM);
  bcomb_kernel<<<H3 / 8, 256, 0, stream>>>(Whh0, bhh0, bhm0, bc0);
  bcomb_kernel<<<H3 / 8, 256, 0, stream>>>(Whh1, bhh1, bhm1, bc1);

  // ----- layer 0 -----
  dim3 ggi(H3 / 128, S_LEN / 128);
  gemm_staged<0><<<ggi, 256, 0, stream>>>(xbf, wih0b, bih0, gi, S_LEN, H3, IN_DIM);
  bar_init<<<1, 1, 0, stream>>>(bar);
  gru_scan<<<NB_REC, 256, 0, stream>>>(gi, wc0, whm0b, bc0, bhm0, hx,
                                       y0, out + (size_t)S_LEN * H_DIM, hbuf, bar);

  // ----- layer 1 -----
  cvt(y0, y0b, S_LEN * H_DIM);
  gemm_staged<0><<<ggi, 256, 0, stream>>>(y0b, wih1b, bih1, gi, S_LEN, H3, H_DIM);
  bar_init<<<1, 1, 0, stream>>>(bar);
  gru_scan<<<NB_REC, 256, 0, stream>>>(gi, wc1, whm1b, bc1, bhm1, hx + H_DIM,
                                       out, out + (size_t)S_LEN * H_DIM + H_DIM, hbuf, bar);
}
